// GATLayer_57724360458242
// MI455X (gfx1250) — matrix operand order repaired and checked by reading, not compiled, not run on hardware
//
#include <hip/hip_runtime.h>
#include <hip/hip_bf16.h>

typedef __attribute__((ext_vector_type(16))) _Float16 v16h;
typedef __attribute__((ext_vector_type(8)))  _Float16 v8h;
typedef __attribute__((ext_vector_type(8)))  float    v8f;
typedef __attribute__((ext_vector_type(4)))  float    v4f;

#define GB 4       // batch
#define GN 2048    // nodes
#define GI 256     // input dim
#define GH 4       // heads
#define GD 64      // head dim
#define GE 256     // H*D

static __device__ __forceinline__ v16h cat8(v8h lo, v8h hi) {
  return __builtin_shufflevector(lo, hi, 0,1,2,3,4,5,6,7,8,9,10,11,12,13,14,15);
}

// ---------------------------------------------------------------------------
// k0: convert W (H,I,D) f32 -> Wt (H,D,I) f16  and  W_out (E,E) f32 -> f16
// ---------------------------------------------------------------------------
__global__ void k0_convert(const float* __restrict__ W, const float* __restrict__ Wout,
                           _Float16* __restrict__ Wt, _Float16* __restrict__ WoutH) {
  int tid = blockIdx.x * blockDim.x + threadIdx.x;   // 131072 total
  if (tid < GH * GI * GD) {
    int h = tid / (GI * GD);
    int r = tid % (GI * GD);
    int i = r / GD;
    int d = r % GD;
    Wt[(h * GD + d) * GI + i] = (_Float16)W[tid];
  } else {
    int t = tid - GH * GI * GD;
    WoutH[t] = (_Float16)Wout[t];
  }
}

// ---------------------------------------------------------------------------
// k1: projection Hp[b,h,n,d] = X[b] @ W[h], stored transposed HpT[b][h][d][n] f16
// block = 128 threads (4 waves), wave h = head; 16 n-rows per block.
// K-loop force-unrolled so all 64 global_load_b128 get distinct registers and
// the WMMAs run behind a descending s_wait_loadcnt pipeline.
// ---------------------------------------------------------------------------
__global__ void __launch_bounds__(128) k1_proj(const float* __restrict__ X,
                                               const _Float16* __restrict__ Wt,
                                               _Float16* __restrict__ HpT) {
  __shared__ __align__(16) _Float16 XL[16 * GI];   // 8 KB f16 X tile
  const int b     = blockIdx.y;
  const int nbase = blockIdx.x * 16;
  const int tid   = threadIdx.x;

  // stage + convert X tile (16 x 256 f32 -> f16), coalesced
  const float* xsrc = X + ((size_t)b * GN + nbase) * GI;
  for (int q = 0; q < 32; ++q) {
    int idx = q * 128 + tid;
    XL[idx] = (_Float16)xsrc[idx];
  }
  __syncthreads();

  const int h = tid >> 5, lane = tid & 31, lr = lane & 15, lh = lane >> 4;
  v8f acc[4] = {};
#pragma unroll
  for (int ks = 0; ks < 8; ++ks) {                  // K = 256 in steps of 32
    const int k0 = ks * 32;
    v16h afrag = cat8(*(const v8h*)&XL[lr * GI + k0 + lh * 8],
                      *(const v8h*)&XL[lr * GI + k0 + lh * 8 + 16]);
    v16h bfrag[4];
#pragma unroll
    for (int dt = 0; dt < 4; ++dt) {
      const int d = dt * 16 + lr;
      const _Float16* bp = Wt + ((h * GD + d) * GI + k0 + lh * 8);
      bfrag[dt] = cat8(*(const v8h*)bp, *(const v8h*)(bp + 16));
    }
#pragma unroll
    for (int dt = 0; dt < 4; ++dt)
      acc[dt] = __builtin_amdgcn_wmma_f32_16x16x32_f16(false, afrag, false, bfrag[dt],
                                                       (short)0, acc[dt], false, false);
  }
  // C layout: VGPR r holds rows M=r (lanes 0-15) / M=r+8 (lanes 16-31), col = lr
  // -> per lane 8 n-consecutive values for fixed d: one b128 store into HpT[d][n]
#pragma unroll
  for (int dt = 0; dt < 4; ++dt) {
    const int d = dt * 16 + lr;
    v8h p;
#pragma unroll
    for (int r = 0; r < 8; ++r) p[r] = (_Float16)acc[dt][r];
    *(v8h*)&HpT[(((size_t)b * GH + h) * GD + d) * GN + nbase + lh * 8] = p;
  }
}

// ---------------------------------------------------------------------------
// k2: H1[b,h,n] = Hp[b,h,n,:].a1 ; H2 = Hp.a2   (reads HpT, coalesced in n)
// ---------------------------------------------------------------------------
__global__ void k2_scores(const _Float16* __restrict__ HpT, const float* __restrict__ a,
                          float* __restrict__ H1, float* __restrict__ H2) {
  int idx = blockIdx.x * blockDim.x + threadIdx.x;   // (b,h,n) flat, 32768
  int n = idx & (GN - 1);
  int h = (idx >> 11) & (GH - 1);
  int b = idx >> 13;
  const _Float16* hp = HpT + ((size_t)b * GH + h) * GD * GN + n;
  const float* a1 = a + h * 2 * GD;
  const float* a2 = a1 + GD;
  float s1 = 0.f, s2 = 0.f;
#pragma unroll 4
  for (int d = 0; d < GD; ++d) {
    float v = (float)hp[(size_t)d * GN];
    s1 += v * a1[d];
    s2 += v * a2[d];
  }
  H1[idx] = s1;
  H2[idx] = s2;
}

// ---------------------------------------------------------------------------
// k3: fused attention. Block = (b, i-tile of 16), 128 thr / 4 waves (wave = head).
// Per 32-wide j-step: all threads compute masked LeakyReLU scores + softmax over
// the 4 HEADS (masked edges -> uniform 0.25), stage as f16 A-matrix in LDS, then
// each wave runs 4x v_wmma_f32_16x16x32_f16 against its HpT j-slab (L2 resident).
// B-fragments are SOFTWARE-PIPELINED one j-iteration ahead: HpT loads are
// independent of the LDS scores, so next-step loads fly during this step's WMMAs.
// Result written flat (b,h,n,d) f16 == the reference's raw reshape (B,N,H*D).
// ---------------------------------------------------------------------------
__global__ void __launch_bounds__(128) k3_attn(const int* __restrict__ A,
                                               const float* __restrict__ H1,
                                               const float* __restrict__ H2,
                                               const _Float16* __restrict__ HpT,
                                               _Float16* __restrict__ Hpr) {
  __shared__ __align__(16) _Float16 sm[4096];  // scores: [4][16][32]; reuse for C staging
  const int b     = blockIdx.y;
  const int ibase = blockIdx.x * 16;
  const int tid   = threadIdx.x;
  const int h = tid >> 5, lane = tid & 31, lr = lane & 15, lh = lane >> 4;

  // phase-1 assignment: thread owns row i = tid>>3, 4 consecutive jj
  const int it  = tid >> 3;
  const int jj0 = (tid & 7) * 4;
  float h1v[GH];
#pragma unroll
  for (int hh = 0; hh < GH; ++hh)
    h1v[hh] = H1[((size_t)b * GH + hh) * GN + ibase + it];
  const int* Arow = A + ((size_t)b * GN + ibase + it) * GN;
  const float* H2b = H2 + (size_t)b * GH * GN;
  const _Float16* Hphb = HpT + ((size_t)b * GH + h) * GD * GN;

  v8f acc[4] = {};

  // prologue: preload B-fragments for jb = 0
  v16h bcur[4];
#pragma unroll
  for (int dt = 0; dt < 4; ++dt) {
    const _Float16* bp = Hphb + (size_t)(dt * 16 + lr) * GN + lh * 8;
    bcur[dt] = cat8(*(const v8h*)bp, *(const v8h*)(bp + 16));
  }

#pragma unroll 1
  for (int jb = 0; jb < GN; jb += 32) {
    // ---- phase 1: scores + softmax over heads ----
#pragma unroll
    for (int q = 0; q < 4; ++q) {
      const int jj = jj0 + q;
      const int j  = jb + jj;
      float att[GH];
      if (Arow[j] > 0) {
        float s[GH], m = -3.4e38f;
#pragma unroll
        for (int hh = 0; hh < GH; ++hh) {
          float x = h1v[hh] + H2b[hh * GN + j];
          s[hh] = x > 0.f ? x : 0.2f * x;           // LeakyReLU(0.2)
          m = fmaxf(m, s[hh]);
        }
        float sum = 0.f;
#pragma unroll
        for (int hh = 0; hh < GH; ++hh) { att[hh] = __expf(s[hh] - m); sum += att[hh]; }
        float inv = 1.0f / sum;
#pragma unroll
        for (int hh = 0; hh < GH; ++hh) att[hh] *= inv;
      } else {
        // all heads masked to NEG_INF -> softmax over heads = 0.25 each
#pragma unroll
        for (int hh = 0; hh < GH; ++hh) att[hh] = 0.25f;
      }
#pragma unroll
      for (int hh = 0; hh < GH; ++hh)
        sm[hh * 512 + it * 32 + jj] = (_Float16)att[hh];
    }
    if (jb + 32 < GN) __builtin_prefetch(Arow + jb + 32, 0, 0);  // global_prefetch_b8
    __syncthreads();

    // ---- phase 2: issue next-step B loads, then WMMA on current fragments ----
    const int jn = (jb + 32 < GN) ? jb + 32 : 0;   // clamp: epilogue load is redundant
    v16h bnext[4];
#pragma unroll
    for (int dt = 0; dt < 4; ++dt) {
      const _Float16* bp = Hphb + (size_t)(dt * 16 + lr) * GN + jn + lh * 8;
      bnext[dt] = cat8(*(const v8h*)bp, *(const v8h*)(bp + 16));
    }
    v16h afrag = cat8(*(const v8h*)&sm[h * 512 + lr * 32 + lh * 8],
                      *(const v8h*)&sm[h * 512 + lr * 32 + lh * 8 + 16]);
#pragma unroll
    for (int dt = 0; dt < 4; ++dt)
      acc[dt] = __builtin_amdgcn_wmma_f32_16x16x32_f16(false, afrag, false, bcur[dt],
                                                       (short)0, acc[dt], false, false);
#pragma unroll
    for (int dt = 0; dt < 4; ++dt) bcur[dt] = bnext[dt];
    __syncthreads();
  }

  // stage C tiles (f16) in LDS, then coalesced b128 global stores
#pragma unroll
  for (int dt = 0; dt < 4; ++dt) {
    const int d = dt * 16 + lr;
#pragma unroll
    for (int r = 0; r < 8; ++r)
      sm[h * 1024 + (lh * 8 + r) * 64 + d] = (_Float16)acc[dt][r];
  }
  __syncthreads();
  {
    const int idx0 = tid * 32;            // 4096 halves total
    const int h2 = idx0 >> 10;
    const int m2 = (idx0 >> 6) & 15;
    const int d0 = idx0 & 63;
    _Float16* dst = Hpr + (((size_t)b * GH + h2) * GN + ibase + m2) * GD + d0;
#pragma unroll
    for (int u = 0; u < 4; ++u)
      *(v8h*)(dst + u * 8) = *(const v8h*)&sm[idx0 + u * 8];
  }
}

// ---------------------------------------------------------------------------
// k4: out[b,n2,:] = Hpr_flat[b,n2,:] @ W_out^T + b_out   (raw-reshape GEMM)
// block = (n2-tile of 16, b); wave w covers output cols w*64..w*64+63.
// ---------------------------------------------------------------------------
__global__ void __launch_bounds__(128) k4_out(const _Float16* __restrict__ Hpr,
                                              const _Float16* __restrict__ WoutH,
                                              const float* __restrict__ bout,
                                              float* __restrict__ out) {
  __shared__ __align__(16) float smf[16 * GE];   // 16 KB C staging
  const int b     = blockIdx.y;
  const int nbase = blockIdx.x * 16;
  const int tid   = threadIdx.x;
  const int w = tid >> 5, lane = tid & 31, lr = lane & 15, lh = lane >> 4;

  v8f acc[4] = {};
  const _Float16* Abase = Hpr + ((size_t)b * GN + nbase + lr) * GE;
#pragma unroll
  for (int ks = 0; ks < 8; ++ks) {                // K = 256
    const int k0 = ks * 32;
    v16h afrag = cat8(*(const v8h*)(Abase + k0 + lh * 8),
                      *(const v8h*)(Abase + k0 + lh * 8 + 16));
    v16h bfrag[4];
#pragma unroll
    for (int et = 0; et < 4; ++et) {
      const int e = w * 64 + et * 16 + lr;
      // B[k][e] = W_out[e][k]: k contiguous in row-major W_out
      const _Float16* bp = WoutH + (size_t)e * GE + k0 + lh * 8;
      bfrag[et] = cat8(*(const v8h*)bp, *(const v8h*)(bp + 16));
    }
#pragma unroll
    for (int et = 0; et < 4; ++et)
      acc[et] = __builtin_amdgcn_wmma_f32_16x16x32_f16(false, afrag, false, bfrag[et],
                                                       (short)0, acc[et], false, false);
  }
#pragma unroll
  for (int et = 0; et < 4; ++et) {
    const int e = w * 64 + et * 16 + lr;
#pragma unroll
    for (int r = 0; r < 8; ++r) smf[(lh * 8 + r) * GE + e] = acc[et][r];
  }
  __syncthreads();
  const int row = tid >> 3;
  const int c0  = (tid & 7) * 32;
  float* dst = out + ((size_t)b * GN + nbase + row) * GE;
#pragma unroll
  for (int u = 0; u < 8; ++u) {
    const int c = c0 + u * 4;
    v4f v  = *(const v4f*)&smf[row * GE + c];
    v4f bb = *(const v4f*)&bout[c];
    v += bb;
    *(v4f*)(dst + c) = v;
  }
}

// ---------------------------------------------------------------------------
extern "C" void kernel_launch(void* const* d_in, const int* in_sizes, int n_in,
                              void* d_out, int out_size, void* d_ws, size_t ws_size,
                              hipStream_t stream) {
  const float* X    = (const float*)d_in[0];
  const int*   A    = (const int*)  d_in[1];
  const float* W    = (const float*)d_in[2];
  const float* a    = (const float*)d_in[3];
  const float* Wout = (const float*)d_in[4];
  const float* bout = (const float*)d_in[5];
  float* out = (float*)d_out;

  char* ws = (char*)d_ws;
  _Float16* Wt    = (_Float16*)(ws);                                   // 128 KB
  _Float16* WoutH = (_Float16*)(ws + (128 << 10));                     // 128 KB
  _Float16* HpT   = (_Float16*)(ws + (256 << 10));                     // 4 MB
  float*    H1    = (float*)   (ws + (256 << 10) + (4 << 20));         // 128 KB
  float*    H2    = (float*)   (ws + (384 << 10) + (4 << 20));         // 128 KB
  _Float16* Hpr   = (_Float16*)(ws + (512 << 10) + (4 << 20));         // 4 MB

  k0_convert<<<dim3(512),        dim3(256), 0, stream>>>(W, Wout, Wt, WoutH);
  k1_proj  <<<dim3(GN/16, GB),   dim3(128), 0, stream>>>(X, Wt, HpT);
  k2_scores<<<dim3((GB*GH*GN)/256), dim3(256), 0, stream>>>(HpT, a, H1, H2);
  k3_attn  <<<dim3(GN/16, GB),   dim3(128), 0, stream>>>(A, H1, H2, HpT, Hpr);
  k4_out   <<<dim3(GN/16, GB),   dim3(128), 0, stream>>>(Hpr, WoutH, bout, out);
}
